// AttentionLSTM_21019569947035
// MI455X (gfx1250) — matrix operand order repaired and checked by reading, not compiled, not run on hardware
//
#include <hip/hip_runtime.h>
#include <hip/hip_bf16.h>

// Persistent single-WGP attention-LSTM for MI455X (gfx1250, wave32).
// All per-step GEMMs run on v_wmma_f32_16x16x32_f16 (f32 accumulate).
// State: h (f16) in LDS, c (f32) in workspace; weights pre-transposed to
// K-major f16 by a prep kernel. Address spaces are explicit so weights load
// as 16-bit global_load gathers (LOADcnt only) and fragments as ds_load_b128.

#define AS_G __attribute__((address_space(1)))
#define AS_L __attribute__((address_space(3)))

typedef __attribute__((ext_vector_type(16))) _Float16     v16h;
typedef __attribute__((ext_vector_type(8)))  float        v8f;
typedef __attribute__((ext_vector_type(4)))  unsigned int u32x4;

union FragU { v16h v; u32x4 q[2]; };

__device__ inline v8f wmma_f16(v16h a, v16h b, v8f c) {
  return __builtin_amdgcn_wmma_f32_16x16x32_f16(false, a, false, b, (short)0, c, false, false);
}

// A fragment: 16x32 f16 tile, rows m0..m0+15, cols k0..k0+31 of row-major [..][ld].
// ISA layout: lane&15 = M; lane half selects K sub-ranges {0..7,16..23}/{8..15,24..31}.
// Per lane this is two contiguous 16B chunks at +0 and +16 elements.
__device__ inline v16h load_a_frag_lds(const AS_L _Float16* buf, int ld, int m0, int k0) {
  const int lane = threadIdx.x & 31;
  const AS_L _Float16* p = buf + (m0 + (lane & 15)) * ld + k0 + ((lane >> 4) << 3);
  FragU f;
  f.q[0] = *(const AS_L u32x4*)p;
  f.q[1] = *(const AS_L u32x4*)(p + 16);
  return f.v;
}
__device__ inline v16h load_a_frag_glb(const AS_G _Float16* buf, int ld, int m0, int k0) {
  const int lane = threadIdx.x & 31;
  const AS_G _Float16* p = buf + (size_t)(m0 + (lane & 15)) * ld + k0 + ((lane >> 4) << 3);
  FragU f;
  f.q[0] = *(const AS_G u32x4*)p;
  f.q[1] = *(const AS_G u32x4*)(p + 16);
  return f.v;
}

// B fragment: 32x16 tile of K-major weight Wt[K][ld]; lane&15 = N col,
// elems = K {0..7,16..23}/{8..15,24..31} by lane half (strided gather).
__device__ inline v16h load_b_frag_glb(const AS_G _Float16* wt, int ld, int k0, int n0) {
  const int lane = threadIdx.x & 31;
  const AS_G _Float16* p = wt + (size_t)(k0 + ((lane >> 4) << 3)) * ld + n0 + (lane & 15);
  v16h v;
#pragma unroll
  for (int s = 0; s < 8; ++s) {
    v[s] = p[(size_t)s * ld];
    v[8 + s] = p[(size_t)(16 + s) * ld];
  }
  return v;
}

// Locality 3 -> nearest (WGP) scope prefetch: actually fills the WGP cache.
// Unconditional: an over-the-end prefetch lands in the adjacent workspace
// allocation (still mapped) and prefetches have no data side effects.
__device__ inline void pf_glb(const AS_G _Float16* p) {
  __builtin_prefetch((const void*)(const _Float16*)p, 0, 3);
}

__device__ inline float sigf(float x) { return 1.f / (1.f + __expf(-x)); }

// One LSTM layer step for all B=64 rows. 16 waves; each wave handles 4
// (Mtile, h-col-tile) combos; per combo the i/f/g/o accumulator tiles stay in
// registers and the cell update happens in-register. h writes deferred past a
// barrier (WAR vs other waves' fragment reads of h).
__device__ inline void lstm_layer(const AS_L _Float16* xbuf, int xld, int kx,
                                  const AS_L _Float16* hbuf, int hld,
                                  const AS_G _Float16* wxt, const AS_G _Float16* wht,
                                  const AS_G float* b_ih, const AS_G float* b_hh,
                                  AS_G float* cst,
                                  AS_L _Float16* hdst, int hdld,
                                  AS_L _Float16* cdst, int cdld,
                                  AS_G _Float16* hexp) {
  const int wave = threadIdx.x >> 5;          // 0..15
  const int lane = threadIdx.x & 31;
  const int hf = lane >> 4, nl = lane & 15;
  _Float16 hstash[4][8];
  int m0s[4], nhs[4];
#pragma unroll
  for (int i = 0; i < 4; ++i) {
    const int id = wave + (i << 4);           // 0..63 combos
    const int m0 = (id >> 4) << 4;            // M tile base (0..48)
    const int nh0 = (id & 15) << 4;           // h-column tile base (0..240)
    v8f acc[4];
#pragma unroll
    for (int g = 0; g < 4; ++g) acc[g] = (v8f){0.f,0.f,0.f,0.f,0.f,0.f,0.f,0.f};
    for (int k0 = 0; k0 < kx; k0 += 32) {
      v16h a = load_a_frag_lds(xbuf, xld, m0, k0);
      pf_glb(wxt + (size_t)(k0 + 32 + lane) * 1024 + nh0);
#pragma unroll
      for (int g = 0; g < 4; ++g)
        acc[g] = wmma_f16(a, load_b_frag_glb(wxt, 1024, k0, g * 256 + nh0), acc[g]);
    }
    for (int k0 = 0; k0 < 256; k0 += 32) {
      v16h a = load_a_frag_lds(hbuf, hld, m0, k0);
      pf_glb(wht + (size_t)(k0 + 32 + lane) * 1024 + nh0);
#pragma unroll
      for (int g = 0; g < 4; ++g)
        acc[g] = wmma_f16(a, load_b_frag_glb(wht, 1024, k0, g * 256 + nh0), acc[g]);
    }
    const int nh = nh0 + nl;
    const float bi = b_ih[nh]       + b_hh[nh];
    const float bf = b_ih[256 + nh] + b_hh[256 + nh];
    const float bg = b_ih[512 + nh] + b_hh[512 + nh];
    const float bo = b_ih[768 + nh] + b_hh[768 + nh];
#pragma unroll
    for (int r = 0; r < 8; ++r) {
      const int M = m0 + r + (hf << 3);
      const float gi = acc[0][r] + bi;
      const float gf = acc[1][r] + bf;
      const float gg = acc[2][r] + bg;
      const float go = acc[3][r] + bo;
      const float co = cst[M * 256 + nh];
      const float cn = sigf(gf) * co + sigf(gi) * tanhf(gg);
      cst[M * 256 + nh] = cn;                       // per-lane exclusive
      if (cdst) cdst[M * cdld + nh] = (_Float16)cn; // f16 mirror (not read this phase)
      hstash[i][r] = (_Float16)(sigf(go) * tanhf(cn));
    }
    m0s[i] = m0; nhs[i] = nh;
  }
  __syncthreads();   // all fragment reads of h done
#pragma unroll
  for (int i = 0; i < 4; ++i) {
#pragma unroll
    for (int r = 0; r < 8; ++r) {
      const int M = m0s[i] + r + (hf << 3);
      hdst[M * hdld + nhs[i]] = hstash[i][r];
      if (hexp) hexp[M * (64 * 256) + nhs[i]] = hstash[i][r];
    }
  }
  __syncthreads();
}

// out[64][ncols] (f32, LDS) = hc1(64x512 f16) @ Wt(512 x ncols) + bias
__device__ inline void attn_gemm(const AS_L _Float16* abuf, int ald,
                                 const AS_G _Float16* wt, int ncols,
                                 const AS_G float* bias, AS_L float* out) {
  const int wave = threadIdx.x >> 5;
  const int lane = threadIdx.x & 31;
  const int ntiles = ncols >> 4;
  const int tiles = ntiles << 2;        // 4 M-tiles
  const int per = tiles >> 4;           // tiles per wave (1 or 4)
  for (int i = 0; i < per; ++i) {
    const int id = wave * per + i;
    const int mt = id / ntiles, nt = id % ntiles;
    v8f acc = (v8f){0.f,0.f,0.f,0.f,0.f,0.f,0.f,0.f};
    for (int k0 = 0; k0 < 512; k0 += 32)
      acc = wmma_f16(load_a_frag_lds(abuf, ald, mt * 16, k0),
                     load_b_frag_glb(wt, ncols, k0, nt * 16), acc);
    const int hf = lane >> 4, nl = lane & 15;
#pragma unroll
    for (int r = 0; r < 8; ++r)
      out[(mt * 16 + r + (hf << 3)) * ncols + nt * 16 + nl] = acc[r] + bias[nt * 16 + nl];
  }
  __syncthreads();
}

// ---------------- prep kernel: f16 K-major weight packs + Xc ----------------
__global__ void attn_lstm_prep(
    const float* input, const float* w_ih_e0, const float* w_ih_e1,
    const float* w_hh_e, const float* w_ih_d0, const float* w_ih_d1,
    const float* w_hh_d, const float* attn_w1, const float* attnd_w1,
    _Float16* We0t, _Float16* We1t, _Float16* Whhe0t, _Float16* Whhe1t,
    _Float16* Wd0t, _Float16* Wd1t, _Float16* Whhd0t, _Float16* Whhd1t,
    _Float16* W1hct, _Float16* Wdhct, _Float16* Wdhet, float* Xc) {
  const int gtid = blockIdx.x * blockDim.x + threadIdx.x;
  const int gsz = gridDim.x * blockDim.x;
  auto tr = [&](_Float16* dst, const float* src, int N, int K) {
    AS_G _Float16* d = (AS_G _Float16*)dst;
    const AS_G float* s = (const AS_G float*)src;
    for (int i = gtid; i < N * K; i += gsz) {
      const int k = i / N, n = i - k * N;
      d[i] = (_Float16)s[n * K + k];
    }
  };
  tr(We0t,   w_ih_e0,              1024, 64);
  tr(We1t,   w_ih_e1,              1024, 256);
  tr(Whhe0t, w_hh_e,               1024, 256);
  tr(Whhe1t, w_hh_e + 1024 * 256,  1024, 256);
  tr(Wd1t,   w_ih_d1,              1024, 256);
  tr(Whhd0t, w_hh_d,               1024, 256);
  tr(Whhd1t, w_hh_d + 1024 * 256,  1024, 256);
  {
    AS_G _Float16* d = (AS_G _Float16*)Wd0t;
    const AS_G float* s = (const AS_G float*)w_ih_d0;
    for (int i = gtid; i < 288 * 1024; i += gsz) {     // K-padded decoder layer0
      const int k = i >> 10, n = i & 1023;
      d[i] = (k < 257) ? (_Float16)s[n * 257 + k] : (_Float16)0.f;
    }
  }
  {
    AS_G _Float16* d = (AS_G _Float16*)W1hct;
    const AS_G float* s = (const AS_G float*)attn_w1;
    for (int i = gtid; i < 512 * 64; i += gsz) {       // attn_w1[:, :512]^T
      const int k = i >> 6, eo = i & 63;
      d[i] = (_Float16)s[eo * 576 + k];
    }
  }
  {
    AS_G _Float16* d1 = (AS_G _Float16*)Wdhct;
    AS_G _Float16* d2 = (AS_G _Float16*)Wdhet;
    const AS_G float* s = (const AS_G float*)attnd_w1;
    for (int i = gtid; i < 512 * 256; i += gsz) {      // attnd_w1[:, :512]^T
      const int k = i >> 8, o = i & 255;
      d1[i] = (_Float16)s[o * 768 + k];
    }
    for (int i = gtid; i < 256 * 256; i += gsz) {      // attnd_w1[:, 512:768]^T
      const int k = i >> 8, o = i & 255;
      d2[i] = (_Float16)s[o * 768 + 512 + k];
    }
  }
  // Xc[b][e][eo] = sum_t input[b,t,1+e] * attn_w1[eo, 512+t]  (time-invariant)
  {
    const AS_G float* in = (const AS_G float*)input;
    const AS_G float* w1 = (const AS_G float*)attn_w1;
    AS_G float* xc = (AS_G float*)Xc;
    for (int i = gtid; i < 64 * 64 * 64; i += gsz) {
      const int b = i >> 12, e = (i >> 6) & 63, eo = i & 63;
      float s = 0.f;
      for (int t = 0; t < 64; ++t)
        s += in[b * 4160 + t * 65 + 1 + e] * w1[eo * 576 + 512 + t];
      xc[i] = s;
    }
  }
}

// ---------------- persistent main kernel: 1 block, 512 threads -------------
__global__ __launch_bounds__(512) void attn_lstm_main(
    const float* input_,
    const float* b_ih_e_, const float* b_hh_e_,
    const float* b_ih_d_, const float* b_hh_d_,
    const float* attn_b1_, const float* attn_w2_,
    const float* attnd_b1_, const float* attnd_w2_,
    const float* proj_w_, const float* proj_b_,
    const _Float16* We0t_, const _Float16* We1t_,
    const _Float16* Whhe0t_, const _Float16* Whhe1t_,
    const _Float16* Wd0t_, const _Float16* Wd1t_,
    const _Float16* Whhd0t_, const _Float16* Whhd1t_,
    const _Float16* W1hct_, const _Float16* Wdhct_, const _Float16* Wdhet_,
    const float* Xc_, _Float16* hexp_, float* Y_, float* cst_, float* plast_,
    float* out_) {
  extern __shared__ char smem_raw[];
  AS_L _Float16* x0  = (AS_L _Float16*)smem_raw;  // [64][288] layer-0 input (padded)
  AS_L _Float16* h0  = x0 + 64 * 288;             // [64][256] layer-0 h
  AS_L _Float16* hc1 = h0 + 64 * 256;             // [64][512] = [h1 | c1] (attention A)
  AS_L float* hc2 = (AS_L float*)(hc1 + 64 * 512);// [64][256] attention hc (enc: 64x64)
  AS_L float* lg  = hc2 + 64 * 256;               // [64][64] logits / softmax a
  const int tid = threadIdx.x;

  // global-addressspace views of everything hot
  const AS_G float* input    = (const AS_G float*)input_;
  const AS_G float* attn_w2  = (const AS_G float*)attn_w2_;
  const AS_G float* attnd_w2 = (const AS_G float*)attnd_w2_;
  const AS_G float* Xc       = (const AS_G float*)Xc_;
  const AS_G float* Y        = (const AS_G float*)Y_;
  AS_G float* Yw             = (AS_G float*)Y_;
  AS_G _Float16* hexp        = (AS_G _Float16*)hexp_;
  AS_G float* cst            = (AS_G float*)cst_;
  AS_G float* plast          = (AS_G float*)plast_;

  auto zero_states = [&]() {
    for (int i = tid; i < 64 * 288; i += 512) x0[i] = (_Float16)0.f;
    for (int i = tid; i < 64 * 256; i += 512) h0[i] = (_Float16)0.f;
    for (int i = tid; i < 64 * 512; i += 512) hc1[i] = (_Float16)0.f;
    for (int i = tid; i < 2 * 64 * 256; i += 512) cst[i] = 0.f;
    __syncthreads();
  };
  auto softmax64 = [&]() {
    if (tid < 64) {
      AS_L float* row = lg + tid * 64;
      float m = row[0];
      for (int j = 1; j < 64; ++j) m = fmaxf(m, row[j]);
      float s = 0.f;
      for (int j = 0; j < 64; ++j) { const float e = __expf(row[j] - m); row[j] = e; s += e; }
      const float inv = 1.f / s;
      for (int j = 0; j < 64; ++j) row[j] *= inv;
    }
    __syncthreads();
  };

  zero_states();
  // ===================== encoder =====================
  for (int t = 0; t < 64; ++t) {
    attn_gemm(hc1, 512, (const AS_G _Float16*)W1hct_, 64,
              (const AS_G float*)attn_b1_, hc2);           // hc (64x64) + b1
    for (int i = 0; i < 8; ++i) {                          // logits[b][e]
      const int id = tid + (i << 9);
      const int b = id >> 6;
      const AS_G float* xr = Xc + (size_t)id * 64;
      const AS_L float* hr = hc2 + b * 64;
      float s = 0.f;
      for (int eo = 0; eo < 64; ++eo) s += tanhf(hr[eo] + xr[eo]) * attn_w2[eo];
      lg[id] = s;
    }
    __syncthreads();
    softmax64();
    for (int i = 0; i < 8; ++i) {                          // x_att = x_t * a
      const int id = tid + (i << 9);
      const int b = id >> 6, e = id & 63;
      x0[b * 288 + e] = (_Float16)(input[b * 4160 + t * 65 + 1 + e] * lg[id]);
    }
    __syncthreads();
    lstm_layer(x0, 288, 64, h0, 256,
               (const AS_G _Float16*)We0t_, (const AS_G _Float16*)Whhe0t_,
               (const AS_G float*)b_ih_e_, (const AS_G float*)b_hh_e_,
               cst, h0, 256, (AS_L _Float16*)nullptr, 0, (AS_G _Float16*)nullptr);
    lstm_layer(h0, 256, 256, hc1, 512,
               (const AS_G _Float16*)We1t_, (const AS_G _Float16*)Whhe1t_,
               (const AS_G float*)b_ih_e_ + 1024, (const AS_G float*)b_hh_e_ + 1024,
               cst + 64 * 256, hc1, 512, hc1 + 256, 512, hexp + t * 256);
  }
  // ============== Y = h_expanded @ attnd_w1[:,512:768]^T (once) ==============
  {
    const int wave = tid >> 5, lane = tid & 31;
    const int hf = lane >> 4, nl = lane & 15;
    const AS_G _Float16* Wdhet = (const AS_G _Float16*)Wdhet_;
    for (int id = wave; id < 4096; id += 16) {             // 256 Mt x 16 Nt tiles
      const int m0 = (id >> 4) << 4, n0 = (id & 15) << 4;
      v8f acc = (v8f){0.f,0.f,0.f,0.f,0.f,0.f,0.f,0.f};
      for (int k0 = 0; k0 < 256; k0 += 32)
        acc = wmma_f16(load_a_frag_glb(hexp, 256, m0, k0),
                       load_b_frag_glb(Wdhet, 256, k0, n0), acc);
#pragma unroll
      for (int r = 0; r < 8; ++r)
        Yw[(size_t)(m0 + r + (hf << 3)) * 256 + n0 + nl] = acc[r];
    }
  }
  __syncthreads();
  zero_states();
  // ===================== decoder =====================
  for (int t = 0; t < 64; ++t) {
    attn_gemm(hc1, 512, (const AS_G _Float16*)Wdhct_, 256,
              (const AS_G float*)attnd_b1_, hc2);          // hc2 (64x256) + b1
    for (int i = 0; i < 8; ++i) {                          // logits[b][t']
      const int id = tid + (i << 9);
      const int b = id >> 6, tt = id & 63;
      const AS_G float* yr = Y + (size_t)(b * 64 + tt) * 256;
      const AS_L float* hr = hc2 + b * 256;
      float s = 0.f;
      for (int o = 0; o < 256; ++o) s += tanhf(hr[o] + yr[o]) * attnd_w2[o];
      lg[id] = s;
    }
    __syncthreads();
    softmax64();
    for (int i = 0; i < 32; ++i) {                         // partial[b][h]
      const int id = tid + (i << 9);
      const int b = id >> 8, h = id & 255;
      const AS_G _Float16* hr = hexp + (size_t)b * 16384 + h;
      const AS_L float* ar = lg + b * 64;
      float s = 0.f;
      for (int tt = 0; tt < 64; ++tt) s += ar[tt] * (float)hr[tt * 256];
      x0[b * 288 + h] = (_Float16)s;
      if (t == 63) plast[b * 256 + h] = s;
    }
    if (tid < 64) x0[tid * 288 + 256] = (_Float16)input[tid * 4160 + t * 65];
    __syncthreads();
    lstm_layer(x0, 288, 288, h0, 256,
               (const AS_G _Float16*)Wd0t_, (const AS_G _Float16*)Whhd0t_,
               (const AS_G float*)b_ih_d_, (const AS_G float*)b_hh_d_,
               cst, h0, 256, (AS_L _Float16*)nullptr, 0, (AS_G _Float16*)nullptr);
    lstm_layer(h0, 256, 256, hc1, 512,
               (const AS_G _Float16*)Wd1t_, (const AS_G _Float16*)Whhd1t_,
               (const AS_G float*)b_ih_d_ + 1024, (const AS_G float*)b_hh_d_ + 1024,
               cst + 64 * 256, hc1, 512, hc1 + 256, 512, (AS_G _Float16*)nullptr);
  }
  // ===================== output =====================
  if (tid < 64) {
    const AS_G float* proj_w = (const AS_G float*)proj_w_;
    float s = ((const AS_G float*)proj_b_)[0];
    const AS_L _Float16* h1 = hc1 + tid * 512;
    for (int h = 0; h < 256; ++h) s += (float)h1[h] * proj_w[h];
    const AS_G float* pr = plast + tid * 256;
    for (int h = 0; h < 256; ++h) s += pr[h] * proj_w[256 + h];
    ((AS_G float*)out_)[tid] = fabsf(s);
  }
}

extern "C" void kernel_launch(void* const* d_in, const int* in_sizes, int n_in,
                              void* d_out, int out_size, void* d_ws, size_t ws_size,
                              hipStream_t stream) {
  (void)in_sizes; (void)n_in; (void)out_size; (void)ws_size;
  const float* input    = (const float*)d_in[0];
  const float* w_ih_e0  = (const float*)d_in[1];
  const float* w_ih_e1  = (const float*)d_in[2];
  const float* w_hh_e   = (const float*)d_in[3];
  const float* b_ih_e   = (const float*)d_in[4];
  const float* b_hh_e   = (const float*)d_in[5];
  const float* w_ih_d0  = (const float*)d_in[6];
  const float* w_ih_d1  = (const float*)d_in[7];
  const float* w_hh_d   = (const float*)d_in[8];
  const float* b_ih_d   = (const float*)d_in[9];
  const float* b_hh_d   = (const float*)d_in[10];
  const float* attn_w1  = (const float*)d_in[11];
  const float* attn_b1  = (const float*)d_in[12];
  const float* attn_w2  = (const float*)d_in[13];
  const float* attnd_w1 = (const float*)d_in[14];
  const float* attnd_b1 = (const float*)d_in[15];
  const float* attnd_w2 = (const float*)d_in[16];
  const float* proj_w   = (const float*)d_in[17];
  const float* proj_b   = (const float*)d_in[18];

  char* ws = (char*)d_ws;
  size_t off = 0;
  auto carve = [&](size_t bytes) -> void* {
    void* p = ws + off; off += (bytes + 255) & ~(size_t)255; return p;
  };
  _Float16* We0t   = (_Float16*)carve(64  * 1024 * 2);
  _Float16* We1t   = (_Float16*)carve(256 * 1024 * 2);
  _Float16* Whhe0t = (_Float16*)carve(256 * 1024 * 2);
  _Float16* Whhe1t = (_Float16*)carve(256 * 1024 * 2);
  _Float16* Wd0t   = (_Float16*)carve(288 * 1024 * 2);
  _Float16* Wd1t   = (_Float16*)carve(256 * 1024 * 2);
  _Float16* Whhd0t = (_Float16*)carve(256 * 1024 * 2);
  _Float16* Whhd1t = (_Float16*)carve(256 * 1024 * 2);
  _Float16* W1hct  = (_Float16*)carve(512 * 64  * 2);
  _Float16* Wdhct  = (_Float16*)carve(512 * 256 * 2);
  _Float16* Wdhet  = (_Float16*)carve(256 * 256 * 2);
  float*    Xc     = (float*)   carve(64 * 64 * 64 * 4);
  _Float16* hexp   = (_Float16*)carve((size_t)64 * 64 * 256 * 2);
  float*    Y      = (float*)   carve((size_t)4096 * 256 * 4);
  float*    cst    = (float*)   carve(2 * 64 * 256 * 4);
  float*    plast  = (float*)   carve(64 * 256 * 4);

  attn_lstm_prep<<<256, 256, 0, stream>>>(
      input, w_ih_e0, w_ih_e1, w_hh_e, w_ih_d0, w_ih_d1, w_hh_d,
      attn_w1, attnd_w1,
      We0t, We1t, Whhe0t, Whhe1t, Wd0t, Wd1t, Whhd0t, Whhd1t,
      W1hct, Wdhct, Wdhet, Xc);

  const size_t shmem = (size_t)(64 * 288 + 64 * 256 + 64 * 512) * 2 +
                       (size_t)(64 * 256 + 64 * 64) * 4;   // 217088 B < 320 KB WGP LDS
  attn_lstm_main<<<1, 512, shmem, stream>>>(
      input, b_ih_e, b_hh_e, b_ih_d, b_hh_d,
      attn_b1, attn_w2, attnd_b1, attnd_w2, proj_w, proj_b,
      We0t, We1t, Whhe0t, Whhe1t, Wd0t, Wd1t, Whhd0t, Whhd1t,
      W1hct, Wdhct, Wdhet, Xc, hexp, Y, cst, plast, (float*)d_out);
}
